// MoE_12051678233096
// MI455X (gfx1250) — compile-verified, matrix operand order checked by reading, not run on hardware
//
#include <hip/hip_runtime.h>
#include <hip/hip_bf16.h>
#include <stdint.h>

typedef __attribute__((ext_vector_type(16))) __bf16 v16bf;
typedef __attribute__((ext_vector_type(8)))  __bf16 v8bf;
typedef __attribute__((ext_vector_type(4)))  __bf16 v4bf;
typedef __attribute__((ext_vector_type(8)))  float  v8f;

union Frag16 { v16bf v; v8bf h[2]; };

constexpr int IN_DIM  = 1024;
constexpr int PE      = 2048;   // P * E
constexpr int OUT_DIM = 1024;
constexpr int TOPK    = 4;

constexpr int TM  = 128;  // token tile
constexpr int TN  = 64;   // output-col tile (2 groups of 32 for gemm1)
constexpr int TK  = 32;   // K per WMMA step
constexpr int LDA = 40;   // LDS row stride (bf16): 80 B -> conflict-free, 16B-aligned
constexpr int LDB = 40;
constexpr int LDH = 68;   // LDS row stride (f32) for h tile

// ---------------------------------------------------------------- CDNA5 async copy helpers
__device__ __forceinline__ uint32_t lds_off32(const void* p) {
    // generic LDS pointer: low 32 bits = wave-relative LDS byte address
    return (uint32_t)(size_t)p;
}
// INST_OFFSET is applied to BOTH the LDS and the global address (ISA 08 §4.4)
__device__ __forceinline__ void async_b128(uint32_t lds, const void* g, int imm) {
    asm volatile("global_load_async_to_lds_b128 %0, %1, off offset:%2"
                 :: "v"(lds), "v"(g), "n"(imm) : "memory");
}
__device__ __forceinline__ void wait_async0() {
#if __has_builtin(__builtin_amdgcn_s_wait_asynccnt)
    __builtin_amdgcn_s_wait_asynccnt(0);
#else
    asm volatile("s_wait_asynccnt 0x0" ::: "memory");
#endif
}

// ---------------------------------------------------------------- prep kernels
__global__ void __launch_bounds__(256)
convert_x(const float* __restrict__ x, __bf16* __restrict__ xb, int n) {
    int idx = (blockIdx.x * 256 + threadIdx.x) * 8;
    if (idx < n) {
        float4 f0 = *(const float4*)(x + idx);
        float4 f1 = *(const float4*)(x + idx + 4);
        v8bf o;
        o[0] = (__bf16)f0.x; o[1] = (__bf16)f0.y; o[2] = (__bf16)f0.z; o[3] = (__bf16)f0.w;
        o[4] = (__bf16)f1.x; o[5] = (__bf16)f1.y; o[6] = (__bf16)f1.z; o[7] = (__bf16)f1.w;
        *(v8bf*)(xb + idx) = o;
    }
}

// src f32 [R][C] -> dst bf16 [C][R]
__global__ void __launch_bounds__(256)
transpose_w(const float* __restrict__ src, __bf16* __restrict__ dst, int R, int C) {
    __shared__ __bf16 tile[32 * 33];
    const int r0 = blockIdx.y * 32, c0 = blockIdx.x * 32;
    {
        const int tr = threadIdx.x >> 3;          // 0..31 row
        const int tc = (threadIdx.x & 7) * 4;     // 0..28 col
        float4 f = *(const float4*)(src + (size_t)(r0 + tr) * C + c0 + tc);
        tile[(tc + 0) * 33 + tr] = (__bf16)f.x;
        tile[(tc + 1) * 33 + tr] = (__bf16)f.y;
        tile[(tc + 2) * 33 + tr] = (__bf16)f.z;
        tile[(tc + 3) * 33 + tr] = (__bf16)f.w;
    }
    __syncthreads();
    {
        const int wc = threadIdx.x >> 3;          // 0..31 col (dst row)
        const int wr = (threadIdx.x & 7) * 4;     // 0..28 row (dst col)
        v4bf o;
        o[0] = tile[wc * 33 + wr + 0];
        o[1] = tile[wc * 33 + wr + 1];
        o[2] = tile[wc * 33 + wr + 2];
        o[3] = tile[wc * 33 + wr + 3];
        *(v4bf*)(dst + (size_t)(c0 + wc) * R + r0 + wr) = o;
    }
}

// ---------------------------------------------------------------- gemm1 + topk
// A = xb [Ntok][IN_DIM] bf16 ; Bt = w1t [PE][IN_DIM] bf16 (pre-transposed)
__global__ void __launch_bounds__(256)
gemm1_topk(const __bf16* __restrict__ xb, const __bf16* __restrict__ w1t,
           __bf16* __restrict__ z) {
    __shared__ __align__(16) __bf16 lA[2][TM * LDA];
    __shared__ __align__(16) __bf16 lB[2][TN * LDB];
    __shared__ float lH[TM * LDH];

    const int t    = threadIdx.x;
    const int wave = t >> 5;
    const int lane = t & 31;
    const int m0   = blockIdx.y * TM;
    const int n0   = blockIdx.x * TN;

    // wave -> 32x32 output block: 4 waves in M, 2 in N
    const int mb = (wave >> 1) * 32;
    const int nb = (wave & 1) * 32;

    // async staging roles: A rows 64B (2 copies/thread), B rows 64B (1 copy/thread)
    const __bf16* aptr = xb  + (size_t)(m0 + (t >> 1)) * IN_DIM + (t & 1) * 16;
    const __bf16* bptr = w1t + (size_t)(n0 + (t >> 2)) * IN_DIM + (t & 3) * 8;
    uint32_t lAoff[2], lBoff[2];
    lAoff[0] = lds_off32(&lA[0][(t >> 1) * LDA + (t & 1) * 16]);
    lAoff[1] = lds_off32(&lA[1][(t >> 1) * LDA + (t & 1) * 16]);
    lBoff[0] = lds_off32(&lB[0][(t >> 2) * LDB + (t & 3) * 8]);
    lBoff[1] = lds_off32(&lB[1][(t >> 2) * LDB + (t & 3) * 8]);

    // prologue: tile 0 -> buffer 0
    async_b128(lAoff[0], aptr, 0);
    async_b128(lAoff[0], aptr, 16);
    async_b128(lBoff[0], bptr, 0);
    const __bf16* anext = aptr + TK;
    const __bf16* bnext = bptr + TK;

    v8f acc[2][2] = {};
    int buf = 0;

    for (int k0 = 0; k0 < IN_DIM; k0 += TK) {
        wait_async0();
        __syncthreads();
        if (k0 + TK < IN_DIM) {
            async_b128(lAoff[buf ^ 1], anext, 0);
            async_b128(lAoff[buf ^ 1], anext, 16);
            async_b128(lBoff[buf ^ 1], bnext, 0);
            anext += TK;
            bnext += TK;
        }
        // load all 4 fragments, then 4 back-to-back WMMAs
        const int kb = (lane < 16) ? 0 : 8;
        const int l15 = lane & 15;
        Frag16 fa[2], fb[2];
        #pragma unroll
        for (int mi = 0; mi < 2; ++mi) {
            const int ar = mb + mi * 16 + l15;
            fa[mi].h[0] = *(const v8bf*)&lA[buf][ar * LDA + kb];
            fa[mi].h[1] = *(const v8bf*)&lA[buf][ar * LDA + kb + 16];
        }
        #pragma unroll
        for (int ni = 0; ni < 2; ++ni) {
            const int bc = nb + ni * 16 + l15;
            fb[ni].h[0] = *(const v8bf*)&lB[buf][bc * LDB + kb];
            fb[ni].h[1] = *(const v8bf*)&lB[buf][bc * LDB + kb + 16];
        }
        #pragma unroll
        for (int mi = 0; mi < 2; ++mi)
            #pragma unroll
            for (int ni = 0; ni < 2; ++ni)
                acc[mi][ni] = __builtin_amdgcn_wmma_f32_16x16x32_bf16(
                    false, fa[mi].v, false, fb[ni].v, (short)0, acc[mi][ni], false, false);
        buf ^= 1;
    }
    __syncthreads();

    // ---- spill h tile to LDS
    {
        const int mrow = mb + ((lane >> 4) << 3);
        const int ncol = lane & 15;
        #pragma unroll
        for (int mi = 0; mi < 2; ++mi)
            #pragma unroll
            for (int ni = 0; ni < 2; ++ni)
                #pragma unroll
                for (int i = 0; i < 8; ++i)
                    lH[(mrow + mi * 16 + i) * LDH + nb + ni * 16 + ncol] = acc[mi][ni][i];
    }
    __syncthreads();

    // ---- top-4 of 32 per (row, group); 256 threads = 128 rows x 2 groups
    {
        const int r = t >> 1;
        const int g = t & 1;
        const float* h = &lH[r * LDH + g * 32];
        float t0 = -3.402823466e38f, t1 = t0, t2 = t0, t3 = t0;
        #pragma unroll
        for (int j = 0; j < 32; ++j) {
            float v = h[j];
            if (v > t3) {
                if (v > t0)      { t3 = t2; t2 = t1; t1 = t0; t0 = v; }
                else if (v > t1) { t3 = t2; t2 = t1; t1 = v; }
                else if (v > t2) { t3 = t2; t2 = v; }
                else             { t3 = v; }
            }
        }
        const float thr = t3;
        int gcnt = 0;
        #pragma unroll
        for (int j = 0; j < 32; ++j) gcnt += (h[j] > thr) ? 1 : 0;
        int eq_keep = TOPK - gcnt;   // ties at threshold kept in index order

        v8bf ov[4];
        #pragma unroll
        for (int c = 0; c < 4; ++c)
            #pragma unroll
            for (int i = 0; i < 8; ++i) {
                float v = h[c * 8 + i];
                bool keep = (v > thr);
                if (!keep && v == thr && eq_keep > 0) { keep = true; --eq_keep; }
                ov[c][i] = (__bf16)(keep ? v : 0.0f);
            }
        __bf16* zr = z + (size_t)(m0 + r) * PE + n0 + g * 32;
        #pragma unroll
        for (int c = 0; c < 4; ++c) *(v8bf*)(zr + c * 8) = ov[c];
    }
}

// ---------------------------------------------------------------- gemm2
// A = z [Ntok][PE] bf16 ; Bt = w2t [OUT_DIM][PE] bf16 (pre-transposed)
__global__ void __launch_bounds__(256)
gemm2(const __bf16* __restrict__ z, const __bf16* __restrict__ w2t,
      float* __restrict__ out) {
    __shared__ __align__(16) __bf16 lA[2][TM * LDA];
    __shared__ __align__(16) __bf16 lB[2][TN * LDB];

    const int t    = threadIdx.x;
    const int wave = t >> 5;
    const int lane = t & 31;
    const int m0   = blockIdx.y * TM;
    const int n0   = blockIdx.x * TN;

    const int mb = (wave >> 1) * 32;
    const int nb = (wave & 1) * 32;

    const __bf16* aptr = z   + (size_t)(m0 + (t >> 1)) * PE + (t & 1) * 16;
    const __bf16* bptr = w2t + (size_t)(n0 + (t >> 2)) * PE + (t & 3) * 8;
    uint32_t lAoff[2], lBoff[2];
    lAoff[0] = lds_off32(&lA[0][(t >> 1) * LDA + (t & 1) * 16]);
    lAoff[1] = lds_off32(&lA[1][(t >> 1) * LDA + (t & 1) * 16]);
    lBoff[0] = lds_off32(&lB[0][(t >> 2) * LDB + (t & 3) * 8]);
    lBoff[1] = lds_off32(&lB[1][(t >> 2) * LDB + (t & 3) * 8]);

    async_b128(lAoff[0], aptr, 0);
    async_b128(lAoff[0], aptr, 16);
    async_b128(lBoff[0], bptr, 0);
    const __bf16* anext = aptr + TK;
    const __bf16* bnext = bptr + TK;

    v8f acc[2][2] = {};
    int buf = 0;

    for (int k0 = 0; k0 < PE; k0 += TK) {
        wait_async0();
        __syncthreads();
        if (k0 + TK < PE) {
            async_b128(lAoff[buf ^ 1], anext, 0);
            async_b128(lAoff[buf ^ 1], anext, 16);
            async_b128(lBoff[buf ^ 1], bnext, 0);
            anext += TK;
            bnext += TK;
        }
        const int kb = (lane < 16) ? 0 : 8;
        const int l15 = lane & 15;
        Frag16 fa[2], fb[2];
        #pragma unroll
        for (int mi = 0; mi < 2; ++mi) {
            const int ar = mb + mi * 16 + l15;
            fa[mi].h[0] = *(const v8bf*)&lA[buf][ar * LDA + kb];
            fa[mi].h[1] = *(const v8bf*)&lA[buf][ar * LDA + kb + 16];
        }
        #pragma unroll
        for (int ni = 0; ni < 2; ++ni) {
            const int bc = nb + ni * 16 + l15;
            fb[ni].h[0] = *(const v8bf*)&lB[buf][bc * LDB + kb];
            fb[ni].h[1] = *(const v8bf*)&lB[buf][bc * LDB + kb + 16];
        }
        #pragma unroll
        for (int mi = 0; mi < 2; ++mi)
            #pragma unroll
            for (int ni = 0; ni < 2; ++ni)
                acc[mi][ni] = __builtin_amdgcn_wmma_f32_16x16x32_bf16(
                    false, fa[mi].v, false, fb[ni].v, (short)0, acc[mi][ni], false, false);
        buf ^= 1;
    }

    // ---- store out (f32)
    const int mrow = m0 + mb + ((lane >> 4) << 3);
    const int ncol = n0 + nb + (lane & 15);
    #pragma unroll
    for (int mi = 0; mi < 2; ++mi)
        #pragma unroll
        for (int ni = 0; ni < 2; ++ni)
            #pragma unroll
            for (int i = 0; i < 8; ++i)
                out[(size_t)(mrow + mi * 16 + i) * OUT_DIM + ncol + ni * 16] = acc[mi][ni][i];
}

// ---------------------------------------------------------------- launch
extern "C" void kernel_launch(void* const* d_in, const int* in_sizes, int n_in,
                              void* d_out, int out_size, void* d_ws, size_t ws_size,
                              hipStream_t stream) {
    const float* x  = (const float*)d_in[0];
    const float* w1 = (const float*)d_in[1];
    const float* w2 = (const float*)d_in[2];
    float* out = (float*)d_out;

    const int Ntok = in_sizes[0] / IN_DIM;   // 16384

    char*   ws  = (char*)d_ws;
    size_t  off = 0;
    __bf16* zb  = (__bf16*)(ws + off); off += (size_t)Ntok * PE * 2;       // 64 MB
    __bf16* xbf = (__bf16*)(ws + off); off += (size_t)Ntok * IN_DIM * 2;   // 32 MB
    __bf16* w1t = (__bf16*)(ws + off); off += (size_t)IN_DIM * PE * 2;     // 4 MB
    __bf16* w2t = (__bf16*)(ws + off);                                     // 4 MB

    const int nx = Ntok * IN_DIM;
    convert_x<<<nx / (256 * 8), 256, 0, stream>>>(x, xbf, nx);
    // w1 [IN_DIM][PE] -> w1t [PE][IN_DIM]
    transpose_w<<<dim3(PE / 32, IN_DIM / 32), 256, 0, stream>>>(w1, w1t, IN_DIM, PE);
    // w2 [PE][OUT_DIM] -> w2t [OUT_DIM][PE]
    transpose_w<<<dim3(OUT_DIM / 32, PE / 32), 256, 0, stream>>>(w2, w2t, PE, OUT_DIM);

    dim3 g1(PE / TN, Ntok / TM);             // 32 x 128
    gemm1_topk<<<g1, 256, 0, stream>>>(xbf, w1t, zb);

    dim3 g2(OUT_DIM / TN, Ntok / TM);        // 16 x 128
    gemm2<<<g2, 256, 0, stream>>>(zb, w2t, out);
}
